// BaseLayer_25580825214991
// MI455X (gfx1250) — compile-verified, matrix operand order checked by reading, not run on hardware
//
#include <hip/hip_runtime.h>
#include <hip/hip_bf16.h>
#include <math.h>

#define T_TOK 1024
#define D_DIM 1024
#define F_DIM 4096
#define E_NUM 8
#define TILE_M 32
#define MAX_TILES 40
#define LN_EPS 1e-5f

typedef __attribute__((ext_vector_type(16))) __bf16 v16bf;
typedef __attribute__((ext_vector_type(8)))  float v8f;

struct Frag {            // 8 VGPRs viewed as 16 bf16 / 8 dwords
  union { v16bf v; __bf16 h[16]; unsigned u[8]; };
};

__device__ __forceinline__ unsigned short f2bf(float f) {
  union { float f; unsigned u; } c; c.f = f;
  unsigned r = c.u + 0x7FFFu + ((c.u >> 16) & 1u);
  return (unsigned short)(r >> 16);
}

// ---------- workspace layout (byte offsets) ----------
#define WS_COUNTS    0                                   // int[8]
#define WS_CURSORS   64                                  // int[8]
#define WS_NUMTILES  128                                 // int[1]
#define WS_EXPERT    256                                 // int[T]
#define WS_ALPHA     (WS_EXPERT + 4*T_TOK)               // float[T]
#define WS_PERM      (WS_ALPHA + 4*T_TOK)                // int[T]
#define WS_TEXP      (WS_PERM + 4*T_TOK)                 // int[MAX_TILES]
#define WS_TSTART    (WS_TEXP + 4*MAX_TILES)
#define WS_TVALID    (WS_TSTART + 4*MAX_TILES)
#define WS_ZERO      14336                               // 8KB zeroed row (A pad)
#define WS_HN        24576                               // ushort[T*D]       (2 MB)
#define WS_ACT1      (WS_HN + 2*T_TOK*D_DIM)             // ushort[(T+1)*F]   (8.4 MB, +dump row)
#define WS_DUMP      (WS_ACT1 + 2*(T_TOK+1)*F_DIM)       // float[D] dump for padded out-stores

// ---------------------------------------------------------------------
__global__ void init_kernel(int* counts, unsigned* zrow) {
  int tid = threadIdx.x;
  if (tid < E_NUM) counts[tid] = 0;
  for (int i = tid; i < 2048; i += 256) zrow[i] = 0u;   // 8KB zero row
}

__device__ __forceinline__ float block_reduce(float v, float* sbuf) {
  int tid = threadIdx.x;
  sbuf[tid] = v;
  __syncthreads();
  for (int s = 128; s > 0; s >>= 1) {
    if (tid < s) sbuf[tid] += sbuf[tid + s];
    __syncthreads();
  }
  float r = sbuf[0];
  __syncthreads();
  return r;
}

// one block per token: LN stats, centroid scores, argmax route, bf16 LN out
__global__ __launch_bounds__(256) void routing_kernel(
    const float* __restrict__ x, const float* __restrict__ cent,
    const float* __restrict__ gamma, const float* __restrict__ beta,
    unsigned short* __restrict__ hn, int* __restrict__ expertid,
    float* __restrict__ alpha, int* __restrict__ counts)
{
  __shared__ float sbuf[256];
  const int t   = blockIdx.x;
  const int tid = threadIdx.x;
  const float* xr = x + (long)t * D_DIM;

  float xv[4], s = 0.f, sq = 0.f;
#pragma unroll
  for (int i = 0; i < 4; ++i) {
    int d = i * 256 + tid;
    float v = xr[d];
    xv[i] = v; s += v; sq += v * v;
  }
  float dpart[E_NUM];
#pragma unroll
  for (int e = 0; e < E_NUM; ++e) {
    float p = 0.f;
#pragma unroll
    for (int i = 0; i < 4; ++i) {
      int d = i * 256 + tid;
      p += xv[i] * cent[e * D_DIM + d];
    }
    dpart[e] = p;
  }
  float mu  = block_reduce(s,  sbuf) * (1.0f / D_DIM);
  float ex2 = block_reduce(sq, sbuf) * (1.0f / D_DIM);
  float var = ex2 - mu * mu;
  float dots[E_NUM];
#pragma unroll
  for (int e = 0; e < E_NUM; ++e) dots[e] = block_reduce(dpart[e], sbuf);

  int best = 0; float bs = dots[0];
#pragma unroll
  for (int e = 1; e < E_NUM; ++e) if (dots[e] > bs) { bs = dots[e]; best = e; }

  if (tid == 0) {
    expertid[t] = best;
    alpha[t]    = 1.0f / (1.0f + __expf(-bs));
    atomicAdd(&counts[best], 1);
  }
  float rstd = rsqrtf(var + LN_EPS);
  const float* g = gamma + best * D_DIM;
  const float* b = beta  + best * D_DIM;
#pragma unroll
  for (int i = 0; i < 4; ++i) {
    int d = i * 256 + tid;
    float h = (xv[i] - mu) * rstd * g[d] + b[d];
    hn[(long)t * D_DIM + d] = f2bf(h);
  }
}

// serial tiny scan: expert offsets + TILE_M-token tile descriptors
__global__ void scan_kernel(const int* __restrict__ counts, int* cursors,
                            int* numTiles, int* tExp, int* tStart, int* tValid)
{
  if (threadIdx.x != 0 || blockIdx.x != 0) return;
  int off = 0, nt = 0;
  for (int e = 0; e < E_NUM; ++e) {
    int c = counts[e];
    cursors[e] = off;
    for (int j = 0; j < c; j += TILE_M) {
      tExp[nt]   = e;
      tStart[nt] = off + j;
      int v = c - j; if (v > TILE_M) v = TILE_M;
      tValid[nt] = v;
      ++nt;
    }
    off += c;
  }
  *numTiles = nt;
}

__global__ void scatter_kernel(const int* __restrict__ expertid,
                               int* cursors, int* __restrict__ perm)
{
  int t = blockIdx.x * blockDim.x + threadIdx.x;
  if (t >= T_TOK) return;
  int e = expertid[t];
  int pos = atomicAdd(&cursors[e], 1);
  perm[pos] = t;
}

// ---------------------------------------------------------------------
// B fragment: 16 f32 of one weight row -> bf16 (native converts)
__device__ __forceinline__ void loadB(Frag& B, const float* wp) {
  float4 w0 = *(const float4*)(wp);
  float4 w1 = *(const float4*)(wp + 4);
  float4 w2 = *(const float4*)(wp + 16);
  float4 w3 = *(const float4*)(wp + 20);
  B.h[0]  = (__bf16)w0.x; B.h[1]  = (__bf16)w0.y;
  B.h[2]  = (__bf16)w0.z; B.h[3]  = (__bf16)w0.w;
  B.h[4]  = (__bf16)w1.x; B.h[5]  = (__bf16)w1.y;
  B.h[6]  = (__bf16)w1.z; B.h[7]  = (__bf16)w1.w;
  B.h[8]  = (__bf16)w2.x; B.h[9]  = (__bf16)w2.y;
  B.h[10] = (__bf16)w2.z; B.h[11] = (__bf16)w2.w;
  B.h[12] = (__bf16)w3.x; B.h[13] = (__bf16)w3.y;
  B.h[14] = (__bf16)w3.z; B.h[15] = (__bf16)w3.w;
}
__device__ __forceinline__ void loadA(Frag& A, const unsigned short* ap) {
  *(uint4*)&A.u[0] = *(const uint4*)(ap);
  *(uint4*)&A.u[4] = *(const uint4*)(ap + 16);
}

// GEMM1: act1[t,f] = relu(hn[t,:] . W1[e][f,:] + b1[e][f])   (K = D)
// tile = 32 tokens; wave -> 32x16 via 2 WMMA sharing one B fragment
__global__ __launch_bounds__(256) void gemm1_kernel(
    const unsigned short* __restrict__ hn, const float* __restrict__ W1,
    const float* __restrict__ b1, unsigned short* __restrict__ act1,
    const unsigned short* __restrict__ zero_row,
    const int* __restrict__ perm, const int* __restrict__ tExp,
    const int* __restrict__ tStart, const int* __restrict__ tValid,
    const int* __restrict__ numTiles)
{
  const int tile = blockIdx.y;
  if (tile >= *numTiles) return;

  __shared__ int rows[TILE_M];
  if (threadIdx.x < TILE_M) {
    int v = tValid[tile];
    rows[threadIdx.x] = (threadIdx.x < v) ? perm[tStart[tile] + threadIdx.x] : -1;
  }
  __syncthreads();

  const int wave = threadIdx.x >> 5;
  const int lane = threadIdx.x & 31;
  const int half = lane >> 4;       // K half-lane split
  const int l15  = lane & 15;
  const int e    = tExp[tile];
  const int fcol = blockIdx.x * 128 + wave * 16 + l15;

  const int r0 = rows[l15];
  const int r1 = rows[16 + l15];
  const unsigned short* ap0 =
      ((r0 < 0) ? zero_row : hn + (long)r0 * D_DIM) + half * 8;
  const unsigned short* ap1 =
      ((r1 < 0) ? zero_row : hn + (long)r1 * D_DIM) + half * 8;
  const float* wp = W1 + (long)e * F_DIM * D_DIM + (long)fcol * D_DIM + half * 8;

  v8f acc0 = {}, acc1 = {};
  for (int k0 = 0; k0 < D_DIM; k0 += 32) {
    Frag A0, A1, B;
    loadA(A0, ap0 + k0);
    loadA(A1, ap1 + k0);
    loadB(B, wp + k0);
    __builtin_prefetch(wp + k0 + 128, 0, 1);
    acc0 = __builtin_amdgcn_wmma_f32_16x16x32_bf16(
        false, A0.v, false, B.v, (short)0, acc0, false, false);
    acc1 = __builtin_amdgcn_wmma_f32_16x16x32_bf16(
        false, A1.v, false, B.v, (short)0, acc1, false, false);
  }

  const float bias = b1[e * F_DIM + fcol];
#pragma unroll
  for (int v = 0; v < 8; ++v) {
    int m  = half * 8 + v;
    int t0 = rows[m];
    int t1 = rows[16 + m];
    long o0 = (long)(t0 < 0 ? T_TOK : t0) * F_DIM + fcol;  // padded -> dump row
    long o1 = (long)(t1 < 0 ? T_TOK : t1) * F_DIM + fcol;
    float v0 = acc0[v] + bias; v0 = v0 > 0.f ? v0 : 0.f;
    float v1 = acc1[v] + bias; v1 = v1 > 0.f ? v1 : 0.f;
    act1[o0] = f2bf(v0);
    act1[o1] = f2bf(v1);
  }
}

// GEMM2: out[t,d] = x[t,d] + alpha[t]*(act1[t,:] . W2[e][d,:] + b2[e][d])  (K = F)
__global__ __launch_bounds__(256) void gemm2_kernel(
    const unsigned short* __restrict__ act1, const float* __restrict__ W2,
    const float* __restrict__ b2, const float* __restrict__ x,
    const float* __restrict__ alpha, float* __restrict__ out,
    float* __restrict__ dump, const unsigned short* __restrict__ zero_row,
    const int* __restrict__ perm, const int* __restrict__ tExp,
    const int* __restrict__ tStart, const int* __restrict__ tValid,
    const int* __restrict__ numTiles)
{
  const int tile = blockIdx.y;
  if (tile >= *numTiles) return;

  __shared__ int rows[TILE_M];
  if (threadIdx.x < TILE_M) {
    int v = tValid[tile];
    rows[threadIdx.x] = (threadIdx.x < v) ? perm[tStart[tile] + threadIdx.x] : -1;
  }
  __syncthreads();

  const int wave = threadIdx.x >> 5;
  const int lane = threadIdx.x & 31;
  const int half = lane >> 4;
  const int l15  = lane & 15;
  const int e    = tExp[tile];
  const int dcol = blockIdx.x * 128 + wave * 16 + l15;

  const int r0 = rows[l15];
  const int r1 = rows[16 + l15];
  const unsigned short* ap0 =
      ((r0 < 0) ? zero_row : act1 + (long)r0 * F_DIM) + half * 8;
  const unsigned short* ap1 =
      ((r1 < 0) ? zero_row : act1 + (long)r1 * F_DIM) + half * 8;
  const float* wp = W2 + (long)e * D_DIM * F_DIM + (long)dcol * F_DIM + half * 8;

  v8f acc0 = {}, acc1 = {};
  for (int k0 = 0; k0 < F_DIM; k0 += 32) {
    Frag A0, A1, B;
    loadA(A0, ap0 + k0);
    loadA(A1, ap1 + k0);
    loadB(B, wp + k0);
    __builtin_prefetch(wp + k0 + 128, 0, 1);
    acc0 = __builtin_amdgcn_wmma_f32_16x16x32_bf16(
        false, A0.v, false, B.v, (short)0, acc0, false, false);
    acc1 = __builtin_amdgcn_wmma_f32_16x16x32_bf16(
        false, A1.v, false, B.v, (short)0, acc1, false, false);
  }

  const float bias = b2[e * D_DIM + dcol];
#pragma unroll
  for (int v = 0; v < 8; ++v) {
    int m  = half * 8 + v;
    int t0 = rows[m];
    int t1 = rows[16 + m];
    int c0 = t0 < 0 ? 0 : t0;                 // clamped index for loads
    int c1 = t1 < 0 ? 0 : t1;
    float res0 = x[(long)c0 * D_DIM + dcol] + alpha[c0] * (acc0[v] + bias);
    float res1 = x[(long)c1 * D_DIM + dcol] + alpha[c1] * (acc1[v] + bias);
    float* p0 = (t0 < 0) ? (dump + dcol) : (out + (long)t0 * D_DIM + dcol);
    float* p1 = (t1 < 0) ? (dump + dcol) : (out + (long)t1 * D_DIM + dcol);
    *p0 = res0;
    *p1 = res1;
  }
}

// ---------------------------------------------------------------------
extern "C" void kernel_launch(void* const* d_in, const int* in_sizes, int n_in,
                              void* d_out, int out_size, void* d_ws, size_t ws_size,
                              hipStream_t stream) {
  const float* x     = (const float*)d_in[0];
  const float* cent  = (const float*)d_in[1];
  const float* gamma = (const float*)d_in[2];
  const float* beta  = (const float*)d_in[3];
  const float* W1    = (const float*)d_in[4];
  const float* b1    = (const float*)d_in[5];
  const float* W2    = (const float*)d_in[6];
  const float* b2    = (const float*)d_in[7];
  float* out = (float*)d_out;

  char* ws = (char*)d_ws;
  int*   counts   = (int*)  (ws + WS_COUNTS);
  int*   cursors  = (int*)  (ws + WS_CURSORS);
  int*   numTiles = (int*)  (ws + WS_NUMTILES);
  int*   expertid = (int*)  (ws + WS_EXPERT);
  float* alpha    = (float*)(ws + WS_ALPHA);
  int*   perm     = (int*)  (ws + WS_PERM);
  int*   tExp     = (int*)  (ws + WS_TEXP);
  int*   tStart   = (int*)  (ws + WS_TSTART);
  int*   tValid   = (int*)  (ws + WS_TVALID);
  unsigned*       zrow = (unsigned*)      (ws + WS_ZERO);
  unsigned short* hn   = (unsigned short*)(ws + WS_HN);
  unsigned short* act1 = (unsigned short*)(ws + WS_ACT1);
  float*          dump = (float*)         (ws + WS_DUMP);

  init_kernel<<<1, 256, 0, stream>>>(counts, zrow);
  routing_kernel<<<T_TOK, 256, 0, stream>>>(x, cent, gamma, beta, hn,
                                            expertid, alpha, counts);
  scan_kernel<<<1, 32, 0, stream>>>(counts, cursors, numTiles, tExp, tStart, tValid);
  scatter_kernel<<<T_TOK / 256, 256, 0, stream>>>(expertid, cursors, perm);

  gemm1_kernel<<<dim3(F_DIM / 128, MAX_TILES), 256, 0, stream>>>(
      hn, W1, b1, act1, (const unsigned short*)zrow,
      perm, tExp, tStart, tValid, numTiles);
  gemm2_kernel<<<dim3(D_DIM / 128, MAX_TILES), 256, 0, stream>>>(
      act1, W2, b2, x, alpha, out, dump, (const unsigned short*)zrow,
      perm, tExp, tStart, tValid, numTiles);
}
